// Transformer_12429635355262
// MI455X (gfx1250) — matrix operand order repaired and checked by reading, not compiled, not run on hardware
//
#include <hip/hip_runtime.h>
#include <hip/hip_bf16.h>

typedef __attribute__((ext_vector_type(16))) __bf16 v16bf;
typedef __attribute__((ext_vector_type(8)))  __bf16 v8bf;
typedef __attribute__((ext_vector_type(8)))  float  v8f;
using bf16_t = __bf16;

// ---------------------------------------------------------------------------
// Generic batched WMMA GEMM: C[m,n] = epi( sum_k A[m,k]*B[k,n] )
// Assumptions satisfied by all call sites: K % 32 == 0; out-of-range rows/cols
// are handled by clamped (duplicate) loads + guarded epilogue stores; every
// tensor has < 2^31 elements so all addressing is 32-bit.
// Tile: 256x128x32 per block, 256 threads (8 waves), wave tile 64x64
// (4x4 WMMA 16x16x32 -> 16 wmma per 16 ds_load_b128: 1.0 loads/wmma).
// ---------------------------------------------------------------------------
struct GemmP {
  const void* A; const void* B; void* C;
  int sar, sac, sao, sai;   // A row,col,outer-batch,inner-batch strides (elems)
  int sbr, sbc, sbo, sbi;   // B strides
  int scr, scc, sco, sci;   // C strides
  int M, N, K, inner;       // inner = inner batch count (z -> outer*inner)
  const float* scale_m;     // per-row scale  (BN fold), nullable
  const float* bias_m;      // per-row bias,  nullable
  const float* bias_n;      // per-col bias,  nullable
  int act;                  // 1 = exact GELU
  int accum;                // 1 = C += result
};

#define LDT 40   // padded LDS row stride (elements): 80B, 16B-aligned, de-conflicted

// ---- staging load helpers (convert to bf16) -------------------------------
__device__ __forceinline__ void load16c(const float* src, bf16_t* dst) {
  const float4* s4 = (const float4*)src;   // 16B-aligned at all call sites
  #pragma unroll
  for (int q = 0; q < 4; ++q) {
    const float4 v = s4[q];
    dst[q * 4 + 0] = (bf16_t)v.x; dst[q * 4 + 1] = (bf16_t)v.y;
    dst[q * 4 + 2] = (bf16_t)v.z; dst[q * 4 + 3] = (bf16_t)v.w;
  }
}
__device__ __forceinline__ void load16c(const bf16_t* src, bf16_t* dst) {
  *(v8bf*)dst       = *(const v8bf*)src;
  *(v8bf*)(dst + 8) = *(const v8bf*)(src + 8);
}
template <typename T>
__device__ __forceinline__ void load16s(const T* src, int stride, bf16_t* dst) {
  #pragma unroll
  for (int e = 0; e < 16; ++e) dst[e] = (bf16_t)(float)src[e * stride];
}

template <typename TA, typename TB, typename TC>
__global__ __launch_bounds__(256) void gemm_wmma(GemmP p) {
  __shared__ __attribute__((aligned(16))) bf16_t lA[256 * LDT];  // 20.0 KB
  __shared__ __attribute__((aligned(16))) bf16_t lB[128 * LDT];  // 10.0 KB

  const int z  = blockIdx.z;
  const int zo = z / p.inner, zi = z % p.inner;
  const TA* A  = (const TA*)p.A + zo * p.sao + zi * p.sai;
  const TB* Bm = (const TB*)p.B + zo * p.sbo + zi * p.sbi;
  TC*       C  = (TC*)p.C      + zo * p.sco + zi * p.sci;

  const int m0 = blockIdx.y * 256, n0 = blockIdx.x * 128;
  const int tid  = threadIdx.x;
  const int lane = tid & 31, wid = tid >> 5;
  const int wm = (wid & 3) * 64;        // wave row offset in tile (4 waves in M)
  const int wn = (wid >> 2) * 64;       // wave col offset in tile (2 waves in N)
  const int half = lane >> 4, r = lane & 15;

  const bool a_colmajor = (p.sar == 1) && (p.sac != 1);
  const bool b_rowmajor = (p.sbc == 1) && (p.sbr != 1);

  v8f acc[4][4];
  #pragma unroll
  for (int i = 0; i < 4; ++i)
    #pragma unroll
    for (int j = 0; j < 4; ++j) acc[i][j] = (v8f)0.f;

  for (int k0 = 0; k0 < p.K; k0 += 32) {
    // ==== stage A tile (256x32) =============================================
    if (!a_colmajor) {
      // row-major: one thread = one full 32-element K strip of one row
      int arow = m0 + tid; arow = arow < p.M ? arow : p.M - 1;
      const TA* src = A + arow * p.sar + k0 * p.sac;
      bf16_t tmp[32];
      if (p.sac == 1) { load16c(src, tmp); load16c(src + 16, tmp + 16); }
      else            { load16s(src, p.sac, tmp); load16s(src + 16 * p.sac, p.sac, tmp + 16); }
      #pragma unroll
      for (int q = 0; q < 4; ++q)
        *(v8bf*)&lA[tid * LDT + q * 8] = *(const v8bf*)&tmp[q * 8];
      if (k0 + 32 < p.K) __builtin_prefetch((const void*)(src + 32 * p.sac), 0, 0);
    } else {
      // col-major (sar==1): contiguous along M, transpose-scatter into LDS
      const int krow = tid >> 3, mb = (tid & 7) * 32;
      const TA* src = A + (k0 + krow) * p.sac + (m0 + mb);
      bf16_t tmp[32];
      if (m0 + mb + 32 <= p.M) { load16c(src, tmp); load16c(src + 16, tmp + 16); }
      else {
        #pragma unroll
        for (int e = 0; e < 32; ++e) {
          int mm = m0 + mb + e; mm = mm < p.M ? mm : p.M - 1;
          tmp[e] = (bf16_t)(float)A[(k0 + krow) * p.sac + mm];
        }
      }
      #pragma unroll
      for (int e = 0; e < 32; ++e) lA[(mb + e) * LDT + krow] = tmp[e];
      if (k0 + 32 < p.K) __builtin_prefetch((const void*)(src + 32 * p.sac), 0, 0);
    }
    // ==== stage B tile (32x128), stored transposed as NxK ===================
    if (b_rowmajor) {
      // row-major (sbc==1): contiguous along N, transpose-scatter into LDS
      const int krow = tid >> 3, nb = (tid & 7) * 16;
      const TB* src = Bm + (k0 + krow) * p.sbr + (n0 + nb);
      bf16_t tmp[16];
      if (n0 + nb + 16 <= p.N) load16c(src, tmp);
      else {
        #pragma unroll
        for (int e = 0; e < 16; ++e) {
          int nn = n0 + nb + e; nn = nn < p.N ? nn : p.N - 1;
          tmp[e] = (bf16_t)(float)Bm[(k0 + krow) * p.sbr + nn];
        }
      }
      #pragma unroll
      for (int e = 0; e < 16; ++e) lB[(nb + e) * LDT + krow] = tmp[e];
      if (k0 + 32 < p.K) __builtin_prefetch((const void*)(src + 32 * p.sbr), 0, 0);
    } else {
      // col-major in memory (sbr small): contiguous/strided along K
      const int col = tid >> 1, kb = (tid & 1) * 16;
      int bcol = n0 + col; bcol = bcol < p.N ? bcol : p.N - 1;
      const TB* src = Bm + (k0 + kb) * p.sbr + bcol * p.sbc;
      bf16_t tmp[16];
      if (p.sbr == 1) load16c(src, tmp);
      else            load16s(src, p.sbr, tmp);
      *(v8bf*)&lB[col * LDT + kb]     = *(const v8bf*)&tmp[0];
      *(v8bf*)&lB[col * LDT + kb + 8] = *(const v8bf*)&tmp[8];
      if (k0 + 32 < p.K) __builtin_prefetch((const void*)(src + 32 * p.sbr), 0, 0);
    }
    __syncthreads();

    // ==== fragments per ISA 7.12.2 (wave32 bf16 16x16x32) ==================
    v16bf af[4], bfr[4];
    #pragma unroll
    for (int j = 0; j < 4; ++j) {
      const bf16_t* pb = &lB[(wn + j * 16 + r) * LDT + half * 8];
      ((v8bf*)&bfr[j])[0] = *(const v8bf*)(pb);
      ((v8bf*)&bfr[j])[1] = *(const v8bf*)(pb + 16);
    }
    #pragma unroll
    for (int i = 0; i < 4; ++i) {
      const bf16_t* pa = &lA[(wm + i * 16 + r) * LDT];
      ((v8bf*)&af[i])[0] = *(const v8bf*)(pa + half * 8);        // K 0..7 / 8..15
      ((v8bf*)&af[i])[1] = *(const v8bf*)(pa + 16 + half * 8);   // K 16..23 / 24..31
    }
    #pragma unroll
    for (int i = 0; i < 4; ++i)
      #pragma unroll
      for (int j = 0; j < 4; ++j)
        acc[i][j] = __builtin_amdgcn_wmma_f32_16x16x32_bf16(
            false, af[i], false, bfr[j], (short)0, acc[i][j], false, false);
    __syncthreads();
  }

  // ---- epilogue: affine / bias / GELU / residual-accumulate ----------------
  #pragma unroll
  for (int i = 0; i < 4; ++i) {
    #pragma unroll
    for (int j = 0; j < 4; ++j) {
      #pragma unroll
      for (int e = 0; e < 8; ++e) {
        const int m = m0 + wm + i * 16 + half * 8 + e;
        const int n = n0 + wn + j * 16 + r;
        if (m < p.M && n < p.N) {
          float v = acc[i][j][e];
          if (p.scale_m) v *= p.scale_m[m];
          if (p.bias_m)  v += p.bias_m[m];
          if (p.bias_n)  v += p.bias_n[n];
          if (p.act == 1) v = 0.5f * v * (1.f + erff(v * 0.7071067811865476f));
          const int off = m * p.scr + n * p.scc;
          if (p.accum) v += (float)C[off];
          C[off] = (TC)v;
        }
      }
    }
  }
}

// ---------------------------------------------------------------------------
// LayerNorm over last dim (512), one wave per row (wave32 shuffle reduction)
// ---------------------------------------------------------------------------
template <typename TO>
__global__ __launch_bounds__(256) void ln_kernel(const float* __restrict__ x,
                                                 const float* __restrict__ w,
                                                 const float* __restrict__ b,
                                                 TO* __restrict__ out, int rows) {
  const int lane = threadIdx.x & 31;
  const int row  = blockIdx.x * 8 + (threadIdx.x >> 5);
  if (row >= rows) return;
  const float* px = x + row * 512;
  float v[16], s = 0.f;
  #pragma unroll
  for (int e = 0; e < 16; ++e) { v[e] = px[lane + e * 32]; s += v[e]; }
  #pragma unroll
  for (int m = 16; m >= 1; m >>= 1) s += __shfl_xor(s, m, 32);
  const float mu = s * (1.f / 512.f);
  float q = 0.f;
  #pragma unroll
  for (int e = 0; e < 16; ++e) { const float d = v[e] - mu; q += d * d; }
  #pragma unroll
  for (int m = 16; m >= 1; m >>= 1) q += __shfl_xor(q, m, 32);
  const float rs = rsqrtf(q * (1.f / 512.f) + 1e-5f);
  #pragma unroll
  for (int e = 0; e < 16; ++e) {
    const int c = lane + e * 32;
    out[row * 512 + c] = (TO)((v[e] - mu) * rs * w[c] + b[c]);
  }
}

// ---------------------------------------------------------------------------
// Row softmax (scaled), bf16 in -> bf16 out, one 256-thread block per row
// ---------------------------------------------------------------------------
__global__ __launch_bounds__(256) void softmax_kernel(const bf16_t* __restrict__ in,
                                                      bf16_t* __restrict__ out,
                                                      int L, float scale) {
  __shared__ float red[8];
  const long long row = blockIdx.x;
  const bf16_t* p = in + row * L;
  const int tid = threadIdx.x, lane = tid & 31, wid = tid >> 5;
  const int cnt = L >> 8;  // 2 or 4
  float vv[4];
  float mx = -3.4e38f;
  for (int i = 0; i < cnt; ++i) { vv[i] = (float)p[tid + i * 256] * scale; mx = fmaxf(mx, vv[i]); }
  #pragma unroll
  for (int m = 16; m >= 1; m >>= 1) mx = fmaxf(mx, __shfl_xor(mx, m, 32));
  if (lane == 0) red[wid] = mx;
  __syncthreads();
  mx = red[0];
  #pragma unroll
  for (int i = 1; i < 8; ++i) mx = fmaxf(mx, red[i]);
  __syncthreads();
  float s = 0.f;
  for (int i = 0; i < cnt; ++i) { vv[i] = __expf(vv[i] - mx); s += vv[i]; }
  #pragma unroll
  for (int m = 16; m >= 1; m >>= 1) s += __shfl_xor(s, m, 32);
  if (lane == 0) red[wid] = s;
  __syncthreads();
  s = 0.f;
  #pragma unroll
  for (int i = 0; i < 8; ++i) s += red[i];
  const float inv = 1.f / s;
  for (int i = 0; i < cnt; ++i) out[row * L + tid + i * 256] = (bf16_t)(vv[i] * inv);
}

// ---------------------------------------------------------------------------
// Channel-mixing 3x3 'SAME' conv over square maps (heads-as-channels / spec)
// grid.y = b*Cc + co  (weights become scalar loads per block)
// ---------------------------------------------------------------------------
__global__ __launch_bounds__(256) void conv3x3_kernel(const bf16_t* __restrict__ in,
                                                      const float* __restrict__ w,
                                                      const float* __restrict__ bias,
                                                      bf16_t* __restrict__ out,
                                                      int Cc, int S) {
  const int bc = blockIdx.y;
  const int b  = bc / Cc, co = bc % Cc;
  const long long pix = (long long)blockIdx.x * 256 + threadIdx.x;
  if (pix >= (long long)S * S) return;
  const int y = (int)(pix / S), x = (int)(pix % S);
  const float* wv = w + (long long)co * Cc * 9;
  float acc = bias[co];
  for (int ci = 0; ci < Cc; ++ci) {
    const bf16_t* pi = in + (long long)(b * Cc + ci) * S * S;
    #pragma unroll
    for (int ky = 0; ky < 3; ++ky) {
      const int yy = y + ky - 1;
      if (yy < 0 || yy >= S) continue;
      #pragma unroll
      for (int kx = 0; kx < 3; ++kx) {
        const int xx = x + kx - 1;
        if (xx < 0 || xx >= S) continue;
        acc += wv[ci * 9 + ky * 3 + kx] * (float)pi[(long long)yy * S + xx];
      }
    }
  }
  out[((long long)(b * Cc + co) * S + y) * S + x] = (bf16_t)acc;
}

// ---------------------------------------------------------------------------
// Depthwise 3x3 over the 32x32 token image, h2 (b,n,c) f32 -> t1 (b,c,p) bf16
// grid.y = b*512 + c
// ---------------------------------------------------------------------------
__global__ __launch_bounds__(256) void dwconv_kernel(const float* __restrict__ h2,
                                                     const float* __restrict__ w,
                                                     const float* __restrict__ bias,
                                                     bf16_t* __restrict__ out) {
  const int bc = blockIdx.y;
  const int b = bc >> 9, c = bc & 511;
  const int p = blockIdx.x * 256 + threadIdx.x;   // 0..1023
  const int y = p >> 5, x = p & 31;
  const float* wv = w + c * 9;
  float acc = bias[c];
  #pragma unroll
  for (int ky = 0; ky < 3; ++ky) {
    const int yy = y + ky - 1;
    if (yy < 0 || yy >= 32) continue;
    #pragma unroll
    for (int kx = 0; kx < 3; ++kx) {
      const int xx = x + kx - 1;
      if (xx < 0 || xx >= 32) continue;
      acc += wv[ky * 3 + kx] * h2[(b * 1024 + (yy * 32 + xx)) * 512 + c];
    }
  }
  out[bc * 1024 + p] = (bf16_t)acc;
}

// Fold BatchNorm(eval) + pointwise bias into per-row affine for the GEMM epilogue
__global__ void bnprep_kernel(const float* g, const float* bb, const float* m,
                              const float* v, const float* pwb,
                              float* alpha, float* beta) {
  const int i = threadIdx.x;  // 256
  const float a = g[i] * rsqrtf(v[i] + 1e-5f);
  alpha[i] = a;
  beta[i]  = (pwb[i] - m[i]) * a + bb[i];
}

// x += h2 + t4^T   (t4 is (b,c,n) f32)
__global__ __launch_bounds__(256) void ffadd_kernel(float* __restrict__ x,
                                                    const float* __restrict__ h2,
                                                    const float* __restrict__ t4) {
  const int i = blockIdx.x * 256 + threadIdx.x;  // over B*N*DIM = 2M
  const int c = i & 511;
  const int bn = i >> 9;
  const int b = bn >> 10, n = bn & 1023;
  x[i] += h2[i] + t4[((b << 9) + c) * 1024 + n];
}

// ---------------------------------------------------------------------------
// Host-side GEMM dispatcher
// ---------------------------------------------------------------------------
template <typename TA, typename TB, typename TC>
static void gemm(hipStream_t s, const void* A, const void* B, void* C,
                 int M, int N, int K, int batches, int inner,
                 int sar, int sac, int sao, int sai,
                 int sbr, int sbc, int sbo, int sbi,
                 int scr, int scc, int sco, int sci,
                 const float* scale_m, const float* bias_m, const float* bias_n,
                 int act, int accum) {
  GemmP p;
  p.A = A; p.B = B; p.C = C;
  p.sar = sar; p.sac = sac; p.sao = sao; p.sai = sai;
  p.sbr = sbr; p.sbc = sbc; p.sbo = sbo; p.sbi = sbi;
  p.scr = scr; p.scc = scc; p.sco = sco; p.sci = sci;
  p.M = M; p.N = N; p.K = K; p.inner = inner;
  p.scale_m = scale_m; p.bias_m = bias_m; p.bias_n = bias_n;
  p.act = act; p.accum = accum;
  dim3 g((N + 127) / 128, (M + 255) / 256, batches);
  gemm_wmma<TA, TB, TC><<<g, dim3(256), 0, s>>>(p);
}

extern "C" void kernel_launch(void* const* d_in, const int* in_sizes, int n_in,
                              void* d_out, int out_size, void* d_ws, size_t ws_size,
                              hipStream_t stream) {
  const int B_ = 4, N_ = 1024, DIM = 512, HEADS = 8, INNER = 512;
  const float SCALE = 0.125f;  // 64^-0.5
  const long long BN = (long long)B_ * N_;           // 4096 tokens

  const float* x_in  = (const float*)d_in[0];
  const float* ln1w  = (const float*)d_in[1];
  const float* ln1b  = (const float*)d_in[2];
  const float* Wqkv  = (const float*)d_in[3];
  const float* Wout  = (const float*)d_in[4];
  const float* bout  = (const float*)d_in[5];
  const float* sp_w  = (const float*)d_in[6];
  const float* sp_b  = (const float*)d_in[7];
  const float* Wqs   = (const float*)d_in[8];
  const float* spec_w= (const float*)d_in[9];
  const float* spec_b= (const float*)d_in[10];
  const float* ln2w  = (const float*)d_in[11];
  const float* ln2b  = (const float*)d_in[12];
  const float* dw_w  = (const float*)d_in[13];
  const float* dw_b  = (const float*)d_in[14];
  const float* pw_w  = (const float*)d_in[15];
  const float* pw_b  = (const float*)d_in[16];
  const float* bn_g  = (const float*)d_in[17];
  const float* bn_b  = (const float*)d_in[18];
  const float* bn_m  = (const float*)d_in[19];
  const float* bn_v  = (const float*)d_in[20];
  const float* c1_w  = (const float*)d_in[21];
  const float* c1_b  = (const float*)d_in[22];
  const float* c2_w  = (const float*)d_in[23];
  const float* c2_b  = (const float*)d_in[24];

  // ---- workspace carve-up (aliasing the big attention-map buffers) ---------
  char* ws = (char*)d_ws;
  size_t off = 0;
  auto alloc = [&](size_t bytes) -> void* {
    void* p = ws + off;
    off += (bytes + 255) & ~(size_t)255;
    return p;
  };
  bf16_t* h_bf      = (bf16_t*)alloc(BN * DIM * 2);                 // LN1 out
  bf16_t* qkv_bf    = (bf16_t*)alloc(BN * 3 * INNER * 2);           // q|k|v
  bf16_t* S_bf      = (bf16_t*)alloc((long long)B_*HEADS*N_*N_*2);  // logits (also reused as P)
  bf16_t* Sm_bf     = (bf16_t*)alloc((long long)B_*HEADS*N_*N_*2);  // softmax maps
  bf16_t* P_bf      = S_bf;                                         // conv'd maps (alias)
  bf16_t* oattn_bf  = (bf16_t*)alloc(BN * INNER * 2);
  bf16_t* osp_bf    = (bf16_t*)alloc(BN * DIM * 2);
  bf16_t* qsks_bf   = (bf16_t*)alloc((long long)B_ * DIM * 2 * N_ * 2);
  bf16_t* S2_bf     = (bf16_t*)alloc((long long)B_ * DIM * DIM * 2);
  bf16_t* S2m_bf    = (bf16_t*)alloc((long long)B_ * DIM * DIM * 2);
  bf16_t* aspec_bf  = S2_bf;                                        // alias
  float*  h2f       = (float*)alloc(BN * DIM * 4);
  bf16_t* t1_bf     = (bf16_t*)alloc((long long)B_ * DIM * 1024 * 2);
  bf16_t* t2_bf     = (bf16_t*)alloc((long long)B_ * 256 * 1024 * 2);
  bf16_t* t3_bf     = (bf16_t*)alloc((long long)B_ * DIM * 1024 * 2);
  float*  t4f       = (float*)alloc((long long)B_ * DIM * 1024 * 4);
  float*  bn_alpha  = (float*)alloc(256 * 4);
  float*  bn_beta   = (float*)alloc(256 * 4);

  // x lives in d_out, updated in place each layer
  float* x = (float*)d_out;
  hipMemcpyAsync(x, x_in, BN * DIM * sizeof(float), hipMemcpyDeviceToDevice, stream);

  for (int l = 0; l < 4; ++l) {
    const float* Wqkv_l = Wqkv + (long long)l * DIM * 3 * INNER;
    const float* Wout_l = Wout + (long long)l * INNER * DIM;
    const float* Wqs_l  = Wqs  + (long long)l * N_ * 3 * N_;

    // 1) h = LN1(x)  -> bf16
    ln_kernel<bf16_t><<<dim3(512), dim3(256), 0, stream>>>(
        x, ln1w + l * DIM, ln1b + l * DIM, h_bf, (int)BN);

    // 2) qkv = h @ Wqkv           (M=4096, K=512, N=1536)
    gemm<bf16_t, float, bf16_t>(stream, h_bf, Wqkv_l, qkv_bf,
        (int)BN, 3 * INNER, DIM, 1, 1,
        DIM, 1, 0, 0,           /*A*/
        3 * INNER, 1, 0, 0,     /*B*/
        3 * INNER, 1, 0, 0,     /*C*/
        nullptr, nullptr, nullptr, 0, 0);

    // 3) S = q @ k^T   per (b,h)  (M=N=1024, K=64, batch=32)
    gemm<bf16_t, bf16_t, bf16_t>(stream, qkv_bf, qkv_bf + INNER, S_bf,
        N_, N_, 64, B_ * HEADS, HEADS,
        3 * INNER, 1, N_ * 3 * INNER, 64,
        1, 3 * INNER, N_ * 3 * INNER, 64,
        N_, 1, HEADS * N_ * N_, N_ * N_,
        nullptr, nullptr, nullptr, 0, 0);

    // 4) softmax rows (scaled)
    softmax_kernel<<<dim3(B_ * HEADS * N_), dim3(256), 0, stream>>>(
        S_bf, Sm_bf, N_, SCALE);

    // 5) head-mixing 3x3 conv over attention maps
    conv3x3_kernel<<<dim3((N_ * N_) / 256, B_ * HEADS), dim3(256), 0, stream>>>(
        Sm_bf, sp_w + (long long)l * HEADS * HEADS * 9, sp_b + l * HEADS,
        P_bf, HEADS, N_);

    // 6) out_h = P @ v  per (b,h) -> packed (b,n,h*64)
    gemm<bf16_t, bf16_t, bf16_t>(stream, P_bf, qkv_bf + 2 * INNER, oattn_bf,
        N_, 64, N_, B_ * HEADS, HEADS,
        N_, 1, HEADS * N_ * N_, N_ * N_,
        3 * INNER, 1, N_ * 3 * INNER, 64,
        INNER, 1, N_ * INNER, 64,
        nullptr, nullptr, nullptr, 0, 0);

    // 7) out_sp = out @ Wout + bout   (M=4096, K=512, N=512)
    gemm<bf16_t, float, bf16_t>(stream, oattn_bf, Wout_l, osp_bf,
        (int)BN, DIM, INNER, 1, 1,
        INNER, 1, 0, 0,
        DIM, 1, 0, 0,
        DIM, 1, 0, 0,
        nullptr, nullptr, bout + l * DIM, 0, 0);

    // 8) [qs|ks] = h^T @ Wqs[:, :2N]  per b (M=512, K=1024, N=2048), A col-major
    gemm<bf16_t, float, bf16_t>(stream, h_bf, Wqs_l, qsks_bf,
        DIM, 2 * N_, N_, B_, 1,
        1, DIM, N_ * DIM, 0,
        3 * N_, 1, 0, 0,
        2 * N_, 1, DIM * 2 * N_, 0,
        nullptr, nullptr, nullptr, 0, 0);

    // 9) S2 = qs @ ks^T  per b (M=N=512, K=1024)
    gemm<bf16_t, bf16_t, bf16_t>(stream, qsks_bf, qsks_bf + N_, S2_bf,
        DIM, DIM, N_, B_, 1,
        2 * N_, 1, DIM * 2 * N_, 0,
        1, 2 * N_, DIM * 2 * N_, 0,
        DIM, 1, DIM * DIM, 0,
        nullptr, nullptr, nullptr, 0, 0);

    // 10) softmax rows (scaled)
    softmax_kernel<<<dim3(B_ * DIM), dim3(256), 0, stream>>>(
        S2_bf, S2m_bf, DIM, SCALE);

    // 11) single-channel 3x3 conv over (dim,dim) map
    conv3x3_kernel<<<dim3((DIM * DIM) / 256, B_), dim3(256), 0, stream>>>(
        S2m_bf, spec_w + (long long)l * 9, spec_b + l, aspec_bf, 1, DIM);

    // 12) x += out_sp @ a_spec  per b (M=1024, K=512, N=512), residual-accumulate
    gemm<bf16_t, bf16_t, float>(stream, osp_bf, aspec_bf, x,
        N_, DIM, DIM, B_, 1,
        DIM, 1, N_ * DIM, 0,
        DIM, 1, DIM * DIM, 0,
        DIM, 1, N_ * DIM, 0,
        nullptr, nullptr, nullptr, 0, 1);

    // 13) h2 = LN2(x) -> f32 (needed for FF internal residual + dw conv)
    ln_kernel<float><<<dim3(512), dim3(256), 0, stream>>>(
        x, ln2w + l * DIM, ln2b + l * DIM, h2f, (int)BN);

    // 14) depthwise 3x3 over 32x32 image -> t1 (b,c,p) bf16
    dwconv_kernel<<<dim3(4, B_ * DIM), dim3(256), 0, stream>>>(
        h2f, dw_w + (long long)l * DIM * 9, dw_b + l * DIM, t1_bf);

    // 15) fold BN+pw_b into per-row affine
    bnprep_kernel<<<dim3(1), dim3(256), 0, stream>>>(
        bn_g + l * 256, bn_b + l * 256, bn_m + l * 256, bn_v + l * 256,
        pw_b + l * 256, bn_alpha, bn_beta);

    // 16) t2 = BN(pw @ t1 + pwb)  per b (M=256, K=512, N=1024)
    gemm<float, bf16_t, bf16_t>(stream, pw_w + (long long)l * 256 * DIM, t1_bf, t2_bf,
        256, 1024, DIM, B_, 1,
        DIM, 1, 0, 0,
        1024, 1, DIM * 1024, 0,
        1024, 1, 256 * 1024, 0,
        bn_alpha, bn_beta, nullptr, 0, 0);

    // 17) t3 = gelu(c1 @ t2 + c1b)  per b (M=512, K=256, N=1024)
    gemm<float, bf16_t, bf16_t>(stream, c1_w + (long long)l * DIM * 256, t2_bf, t3_bf,
        DIM, 1024, 256, B_, 1,
        256, 1, 0, 0,
        1024, 1, 256 * 1024, 0,
        1024, 1, DIM * 1024, 0,
        nullptr, c1_b + l * DIM, nullptr, 1, 0);

    // 18) t4 = gelu(c2 @ t3 + c2b)  per b (M=512, K=512, N=1024) -> f32
    gemm<float, bf16_t, float>(stream, c2_w + (long long)l * DIM * DIM, t3_bf, t4f,
        DIM, 1024, DIM, B_, 1,
        DIM, 1, 0, 0,
        1024, 1, DIM * 1024, 0,
        1024, 1, DIM * 1024, 0,
        nullptr, c2_b + l * DIM, nullptr, 1, 0);

    // 19) x += h2 + t4^T
    ffadd_kernel<<<dim3((unsigned)(BN * DIM / 256)), dim3(256), 0, stream>>>(
        x, h2f, t4f);
  }
  (void)in_sizes; (void)n_in; (void)out_size; (void)ws_size;
}
